// QuantizedAttention_23922967839227
// MI455X (gfx1250) — matrix operand order repaired and checked by reading, not compiled, not run on hardware
//
#include <hip/hip_runtime.h>

typedef __attribute__((ext_vector_type(16))) _Float16 v16h;
typedef __attribute__((ext_vector_type(8)))  _Float16 v8h;
typedef __attribute__((ext_vector_type(8)))  float    v8f;

namespace {
constexpr int B  = 2;
constexpr int S  = 2048;
constexpr int D  = 1024;
constexpr int H  = 16;
constexpr int HD = 64;
constexpr int TOK   = B * S;     // 4096
constexpr int N_QKV = 3 * D;     // 3072
constexpr float SM_SCALE = 0.125f;   // 1/sqrt(64)
constexpr float NEG_BIG  = -1e9f;
}

// ---------------- fragment loaders (contiguous k order, same for A and B) ----------
// A 16x32 f16: lane<16 -> row=lane, K = {0..15}; lane>=16 -> row=lane-16,
// K = {16..31}.  Caller passes p already offset by (hf*16) within the k-chunk.
__device__ __forceinline__ v16h load_frag_a(const _Float16* p) {
  v8h lo = *(const v8h*)(p);
  v8h hi = *(const v8h*)(p + 8);
  v16h r;
#pragma unroll
  for (int j = 0; j < 8; ++j) { r[j] = lo[j]; r[j + 8] = hi[j]; }
  return r;
}
// B 32x16 f16: lane holds column n=lane%16, 16 contiguous K values starting at
// (lane<16 ? 0 : 16).  Caller passes p already offset accordingly.
__device__ __forceinline__ v16h load_frag_b(const _Float16* p) {
  v8h lo = *(const v8h*)(p);
  v8h hi = *(const v8h*)(p + 8);
  v16h r;
#pragma unroll
  for (int j = 0; j < 8; ++j) { r[j] = lo[j]; r[j + 8] = hi[j]; }
  return r;
}

// reductions across the 16 lanes of a half-wave (xor masks stay within half)
__device__ __forceinline__ float hmax16(float v) {
  v = fmaxf(v, __shfl_xor(v, 1));
  v = fmaxf(v, __shfl_xor(v, 2));
  v = fmaxf(v, __shfl_xor(v, 4));
  v = fmaxf(v, __shfl_xor(v, 8));
  return v;
}
__device__ __forceinline__ float hsum16(float v) {
  v += __shfl_xor(v, 1);
  v += __shfl_xor(v, 2);
  v += __shfl_xor(v, 4);
  v += __shfl_xor(v, 8);
  return v;
}

// CDNA5 async global->LDS copy of one 16B chunk per lane (per-lane addresses),
// tracked by ASYNCcnt.  ldsOff = byte offset within the wave's LDS window
// (low 32 bits of a generic pointer to a __shared__ object).
__device__ __forceinline__ void async_copy_b128(unsigned ldsOff, const void* gaddr) {
  unsigned long long ga = (unsigned long long)(uintptr_t)gaddr;
  asm volatile("global_load_async_to_lds_b128 %0, %1, off"
               :: "v"(ldsOff), "v"(ga)
               : "memory");
}
__device__ __forceinline__ void wait_async_le1() {
  asm volatile("s_wait_asynccnt 1" ::: "memory");
}
__device__ __forceinline__ void wait_async_le0() {
  asm volatile("s_wait_asynccnt 0" ::: "memory");
}

// ---------------- per-output-row symmetric weight fake-quant -> f16 ------------
__global__ void k_quant_weight(const float* __restrict__ W, _Float16* __restrict__ Wq,
                               int cols) {
  __shared__ float red[256];
  int row = blockIdx.x;
  const float* wr = W + (size_t)row * cols;
  float m = 0.f;
  for (int c = threadIdx.x; c < cols; c += 256) m = fmaxf(m, fabsf(wr[c]));
  red[threadIdx.x] = m;
  __syncthreads();
  for (int st = 128; st > 0; st >>= 1) {
    if (threadIdx.x < st) red[threadIdx.x] = fmaxf(red[threadIdx.x], red[threadIdx.x + st]);
    __syncthreads();
  }
  float scale = red[0] * (1.0f / 127.0f);
  float inv = (scale > 0.f) ? 1.0f / scale : 0.f;
  _Float16* wo = Wq + (size_t)row * cols;
  for (int c = threadIdx.x; c < cols; c += 256) {
    float q = fminf(fmaxf(rintf(wr[c] * inv), -128.f), 127.f);
    wo[c] = (_Float16)(q * scale);
  }
}

__global__ void k_f32_to_f16(const float* __restrict__ x, _Float16* __restrict__ y, int n) {
  for (int i = blockIdx.x * blockDim.x + threadIdx.x; i < n; i += gridDim.x * blockDim.x)
    y[i] = (_Float16)x[i];
}

// ---------------- cooperative NT GEMM: C[M,N] = A[M,K] * W[N,K]^T + bias ------
// Block = 256 threads (8 waves) computes a 128(M) x 64(N) macro tile.
// The shared 64x32 weight tile per K-step is staged global->LDS with
// global_load_async_to_lds_b128 (double buffered, ASYNCcnt), then each wave
// reads B fragments from LDS (ds_load_b128) and A fragments from global.
__global__ void k_gemm_nt_wmma(const _Float16* __restrict__ A, const _Float16* __restrict__ W,
                               const float* __restrict__ bias, float* __restrict__ C,
                               int M, int N, int K) {
  __shared__ _Float16 Bl[2][64][32];      // 8 KB, double buffered
  int tid = threadIdx.x;
  int wv = tid >> 5, lane = tid & 31;
  int mBlocks = M >> 7;
  int mb = blockIdx.x % mBlocks, nb = blockIdx.x / mBlocks;
  int r = lane & 15, hf = lane >> 4;

  int rowA = mb * 128 + wv * 16 + r;
  const _Float16* ap = A + (size_t)rowA * K + hf * 16;

  // async staging map: thread t copies row (t>>2), 16B chunk (t&3) each K-step
  int brow = tid >> 2, bchunk = tid & 3;
  const _Float16* wsrc = W + (size_t)(nb * 64 + brow) * K + bchunk * 8;
  unsigned ldsOff[2] = {
      (unsigned)(uintptr_t)(&Bl[0][brow][bchunk * 8]),
      (unsigned)(uintptr_t)(&Bl[1][brow][bchunk * 8]) };

  async_copy_b128(ldsOff[0], wsrc);                 // stage 0

  v8f acc[4] = {};
  int steps = K >> 5;
  v16h a_cur = load_frag_a(ap);                     // A frag for step 0

  // steady state: prefetch next stage, compute current (last iter peeled)
  for (int i = 0; i < steps - 1; ++i) {
    int cur = i & 1;
    async_copy_b128(ldsOff[cur ^ 1], wsrc + (size_t)(i + 1) * 32);
    wait_async_le1();                               // own transfers for buf[cur] done
    __syncthreads();                                // buf[cur] complete for all waves
    v16h a_nxt = load_frag_a(ap + (size_t)(i + 1) * 32);
    v16h b0 = load_frag_b(&Bl[cur][ 0 + r][hf * 16]);
    v16h b1 = load_frag_b(&Bl[cur][16 + r][hf * 16]);
    v16h b2 = load_frag_b(&Bl[cur][32 + r][hf * 16]);
    v16h b3 = load_frag_b(&Bl[cur][48 + r][hf * 16]);
    acc[0] = __builtin_amdgcn_wmma_f32_16x16x32_f16(false, a_cur, false, b0, (short)0, acc[0], false, false);
    acc[1] = __builtin_amdgcn_wmma_f32_16x16x32_f16(false, a_cur, false, b1, (short)0, acc[1], false, false);
    acc[2] = __builtin_amdgcn_wmma_f32_16x16x32_f16(false, a_cur, false, b2, (short)0, acc[2], false, false);
    acc[3] = __builtin_amdgcn_wmma_f32_16x16x32_f16(false, a_cur, false, b3, (short)0, acc[3], false, false);
    a_cur = a_nxt;
    __syncthreads();                                // reads of buf[cur] done block-wide
  }
  // final K-step
  {
    int cur = (steps - 1) & 1;
    wait_async_le0();
    __syncthreads();
    v16h b0 = load_frag_b(&Bl[cur][ 0 + r][hf * 16]);
    v16h b1 = load_frag_b(&Bl[cur][16 + r][hf * 16]);
    v16h b2 = load_frag_b(&Bl[cur][32 + r][hf * 16]);
    v16h b3 = load_frag_b(&Bl[cur][48 + r][hf * 16]);
    acc[0] = __builtin_amdgcn_wmma_f32_16x16x32_f16(false, a_cur, false, b0, (short)0, acc[0], false, false);
    acc[1] = __builtin_amdgcn_wmma_f32_16x16x32_f16(false, a_cur, false, b1, (short)0, acc[1], false, false);
    acc[2] = __builtin_amdgcn_wmma_f32_16x16x32_f16(false, a_cur, false, b2, (short)0, acc[2], false, false);
    acc[3] = __builtin_amdgcn_wmma_f32_16x16x32_f16(false, a_cur, false, b3, (short)0, acc[3], false, false);
  }
#pragma unroll
  for (int nn = 0; nn < 4; ++nn) {
    int col = nb * 64 + nn * 16 + r;
    float bv = bias[col];
#pragma unroll
    for (int i = 0; i < 8; ++i) {
      int row = mb * 128 + wv * 16 + i + 8 * hf;
      C[(size_t)row * N + col] = acc[nn][i] + bv;
    }
  }
}

// ---------------- KV amax over (H,S) per (batch, head_dim) --------------------
__global__ void k_kv_amax(const float* __restrict__ qkv, float* __restrict__ kamax,
                          float* __restrict__ vamax) {
  __shared__ float red[256];
  int which = blockIdx.x >> 7;       // 0 = k, 1 = v
  int rem = blockIdx.x & 127;
  int b = rem >> 6, d = rem & 63;
  int off = D + which * D + d;
  float m = 0.f;
  for (int t = threadIdx.x; t < H * S; t += 256) {
    int h = t >> 11, s = t & (S - 1);
    m = fmaxf(m, fabsf(qkv[(size_t)(b * S + s) * N_QKV + off + h * HD]));
  }
  red[threadIdx.x] = m;
  __syncthreads();
  for (int st = 128; st > 0; st >>= 1) {
    if (threadIdx.x < st) red[threadIdx.x] = fmaxf(red[threadIdx.x], red[threadIdx.x + st]);
    __syncthreads();
  }
  if (threadIdx.x == 0) (which ? vamax : kamax)[b * HD + d] = red[0];
}

// ---------------- split heads + KV fake-quant; V stored transposed ------------
__global__ void k_build_heads(const float* __restrict__ qkv,
                              const float* __restrict__ kamax, const float* __restrict__ vamax,
                              _Float16* __restrict__ q16, _Float16* __restrict__ k16,
                              _Float16* __restrict__ vT16,
                              float* __restrict__ kout, float* __restrict__ vout) {
  int n = B * H * S * HD;
  for (int idx = blockIdx.x * blockDim.x + threadIdx.x; idx < n;
       idx += gridDim.x * blockDim.x) {
    int d = idx & 63, s = (idx >> 6) & (S - 1), h = (idx >> 17) & 15, b = idx >> 21;
    size_t base = (size_t)(b * S + s) * N_QKV + h * HD + d;
    size_t hidx = ((size_t)(b * H + h) * S + s) * HD + d;
    q16[hidx] = (_Float16)qkv[base];
    float ks = kamax[b * HD + d] * (1.0f / 127.0f);
    float ki = (ks > 0.f) ? 1.0f / ks : 0.f;
    float kq = fminf(fmaxf(rintf(qkv[base + D] * ki), -128.f), 127.f) * ks;
    kout[hidx] = kq;
    k16[hidx] = (_Float16)kq;
    float vs = vamax[b * HD + d] * (1.0f / 127.0f);
    float vi = (vs > 0.f) ? 1.0f / vs : 0.f;
    float vq = fminf(fmaxf(rintf(qkv[base + 2 * D] * vi), -128.f), 127.f) * vs;
    vout[hidx] = vq;
    vT16[((size_t)(b * H + h) * HD + d) * S + s] = (_Float16)vq;
  }
}

// ---------------- causal flash attention, one wave per 16-query tile ----------
__global__ void k_flash_attn(const _Float16* __restrict__ Q, const _Float16* __restrict__ Kh,
                             const _Float16* __restrict__ Vt, float* __restrict__ O) {
  __shared__ _Float16 P[4][16][32];            // per-wave P-tile staging
  int wid  = (blockIdx.x * blockDim.x + threadIdx.x) >> 5;
  int lane = threadIdx.x & 31;
  int wl   = (threadIdx.x >> 5) & 3;
  int mtiles = S / 16;
  if (wid >= B * H * mtiles) return;
  int b = wid / (H * mtiles);
  int rem = wid % (H * mtiles);
  int h = rem / mtiles;
  int mT = rem % mtiles;
  int m0 = mT * 16;
  int r = lane & 15, hf = lane >> 4;
  const _Float16* Qb = Q  + (size_t)(b * H + h) * S * HD;
  const _Float16* Kb = Kh + (size_t)(b * H + h) * S * HD;
  const _Float16* Vb = Vt + (size_t)(b * H + h) * HD * S;

  v16h qa0 = load_frag_a(Qb + (size_t)(m0 + r) * HD + hf * 16);
  v16h qa1 = load_frag_a(Qb + (size_t)(m0 + r) * HD + 32 + hf * 16);

  v8f o[4] = {};
  float mrun[8], lrun[8];
#pragma unroll
  for (int i = 0; i < 8; ++i) { mrun[i] = -3.0e38f; lrun[i] = 0.f; }

  for (int n0 = 0; n0 <= m0 + 15; n0 += 32) {      // causal: skip fully-masked tiles
    // all four K fragments up-front so DS/VMEM latency pipelines
    v16h kb0 = load_frag_b(Kb + (size_t)(n0 + r) * HD + hf * 16);
    v16h kb1 = load_frag_b(Kb + (size_t)(n0 + r) * HD + 32 + hf * 16);
    v16h kb2 = load_frag_b(Kb + (size_t)(n0 + 16 + r) * HD + hf * 16);
    v16h kb3 = load_frag_b(Kb + (size_t)(n0 + 16 + r) * HD + 32 + hf * 16);
    v8f s0 = {}, s1 = {};
    s0 = __builtin_amdgcn_wmma_f32_16x16x32_f16(false, qa0, false, kb0, (short)0, s0, false, false);
    s0 = __builtin_amdgcn_wmma_f32_16x16x32_f16(false, qa1, false, kb1, (short)0, s0, false, false);
    s1 = __builtin_amdgcn_wmma_f32_16x16x32_f16(false, qa0, false, kb2, (short)0, s1, false, false);
    s1 = __builtin_amdgcn_wmma_f32_16x16x32_f16(false, qa1, false, kb3, (short)0, s1, false, false);

    // issue V fragment loads now: they overlap the whole softmax section below
    v16h vb0 = load_frag_b(Vb + (size_t)( 0 + r) * S + n0 + hf * 16);
    v16h vb1 = load_frag_b(Vb + (size_t)(16 + r) * S + n0 + hf * 16);
    v16h vb2 = load_frag_b(Vb + (size_t)(32 + r) * S + n0 + hf * 16);
    v16h vb3 = load_frag_b(Vb + (size_t)(48 + r) * S + n0 + hf * 16);

    bool needMask = (n0 + 31 > m0 + 8 * hf);       // any column masked for our rows?
#pragma unroll
    for (int i = 0; i < 8; ++i) {
      int qrow = m0 + i + 8 * hf;                  // C-frag row for this VGPR/half
      float a = s0[i] * SM_SCALE;
      float c = s1[i] * SM_SCALE;
      if (needMask) {
        a += ((n0 + r)      <= qrow ? 0.f : NEG_BIG);
        c += ((n0 + 16 + r) <= qrow ? 0.f : NEG_BIG);
      }
      float tm = hmax16(fmaxf(a, c));
      float mn = fmaxf(mrun[i], tm);
      float corr = __expf(mrun[i] - mn);
      float p0 = __expf(a - mn);
      float p1 = __expf(c - mn);
      float rs = hsum16(p0 + p1);
      lrun[i] = lrun[i] * corr + rs;
      mrun[i] = mn;
      o[0][i] *= corr; o[1][i] *= corr; o[2][i] *= corr; o[3][i] *= corr;
      P[wl][i + 8 * hf][r]      = (_Float16)p0;    // C-layout -> LDS
      P[wl][i + 8 * hf][r + 16] = (_Float16)p1;
    }
    asm volatile("" ::: "memory");
    v16h pa;                                       // LDS -> A-layout
#pragma unroll
    for (int j = 0; j < 8; ++j) {
      pa[j]     = P[wl][r][hf * 16 + j];
      pa[j + 8] = P[wl][r][hf * 16 + 8 + j];
    }
    asm volatile("" ::: "memory");
    o[0] = __builtin_amdgcn_wmma_f32_16x16x32_f16(false, pa, false, vb0, (short)0, o[0], false, false);
    o[1] = __builtin_amdgcn_wmma_f32_16x16x32_f16(false, pa, false, vb1, (short)0, o[1], false, false);
    o[2] = __builtin_amdgcn_wmma_f32_16x16x32_f16(false, pa, false, vb2, (short)0, o[2], false, false);
    o[3] = __builtin_amdgcn_wmma_f32_16x16x32_f16(false, pa, false, vb3, (short)0, o[3], false, false);
  }
#pragma unroll
  for (int nhd = 0; nhd < 4; ++nhd)
#pragma unroll
    for (int i = 0; i < 8; ++i) {
      int row = m0 + i + 8 * hf;
      O[(size_t)(b * S + row) * D + h * HD + nhd * 16 + r] = o[nhd][i] / lrun[i];
    }
}

// ---------------- per-batch activation amax (2-pass) + fake-quant -------------
__global__ void k_act_partial(const float* __restrict__ x, float* __restrict__ part) {
  __shared__ float red[256];
  int b = blockIdx.x >> 8, chunk = blockIdx.x & 255;
  const float* xb = x + (size_t)b * (S * D) + chunk * (S * D / 256);
  float m = 0.f;
  for (int t = threadIdx.x; t < S * D / 256; t += 256) m = fmaxf(m, fabsf(xb[t]));
  red[threadIdx.x] = m;
  __syncthreads();
  for (int st = 128; st > 0; st >>= 1) {
    if (threadIdx.x < st) red[threadIdx.x] = fmaxf(red[threadIdx.x], red[threadIdx.x + st]);
    __syncthreads();
  }
  if (threadIdx.x == 0) part[blockIdx.x] = red[0];
}
__global__ void k_act_reduce(const float* __restrict__ part, float* __restrict__ amax) {
  __shared__ float red[256];
  red[threadIdx.x] = part[blockIdx.x * 256 + threadIdx.x];
  __syncthreads();
  for (int st = 128; st > 0; st >>= 1) {
    if (threadIdx.x < st) red[threadIdx.x] = fmaxf(red[threadIdx.x], red[threadIdx.x + st]);
    __syncthreads();
  }
  if (threadIdx.x == 0) amax[blockIdx.x] = red[0];
}
__global__ void k_act_quant(const float* __restrict__ x, const float* __restrict__ amax,
                            _Float16* __restrict__ y) {
  int n = B * S * D;
  for (int i = blockIdx.x * blockDim.x + threadIdx.x; i < n; i += gridDim.x * blockDim.x) {
    int b = i >> 21;                                // S*D = 2^21
    float sc = amax[b] * (1.0f / 127.0f);
    float iv = (sc > 0.f) ? 1.0f / sc : 0.f;
    y[i] = (_Float16)(fminf(fmaxf(rintf(x[i] * iv), -128.f), 127.f) * sc);
  }
}

// ------------------------------- launcher -------------------------------------
extern "C" void kernel_launch(void* const* d_in, const int* in_sizes, int n_in,
                              void* d_out, int out_size, void* d_ws, size_t ws_size,
                              hipStream_t stream) {
  (void)in_sizes; (void)n_in; (void)out_size; (void)ws_size;
  const float* hidden = (const float*)d_in[0];
  const float* w_attn = (const float*)d_in[1];
  const float* b_attn = (const float*)d_in[2];
  const float* w_proj = (const float*)d_in[3];
  const float* b_proj = (const float*)d_in[4];
  // d_in[5] = attention_mask (pure causal; recomputed in-kernel)

  char* ws = (char*)d_ws;
  _Float16* hidden16 = (_Float16*)(ws);                         //  8 MB
  _Float16* wq16     = (_Float16*)(ws + ((size_t) 8 << 20));    //  6 MB
  _Float16* wp16     = (_Float16*)(ws + ((size_t)14 << 20));    //  2 MB
  float*    qkv32    = (float*)   (ws + ((size_t)16 << 20));    // 48 MB
  _Float16* q16      = (_Float16*)(ws + ((size_t)64 << 20));    //  8 MB
  _Float16* k16      = (_Float16*)(ws + ((size_t)72 << 20));    //  8 MB
  _Float16* vT16     = (_Float16*)(ws + ((size_t)80 << 20));    //  8 MB
  float*    attnout  = (float*)   (ws + ((size_t)88 << 20));    // 16 MB
  _Float16* attnq16  = (_Float16*)(ws + ((size_t)104 << 20));   //  8 MB
  float*    kamax    = (float*)   (ws + ((size_t)112 << 20));
  float*    vamax    = kamax + 128;
  float*    part     = vamax + 128;    // 512 entries
  float*    actamax  = part + 512;     // 2 entries

  float* outp  = (float*)d_out;
  float* koutp = outp + (size_t)B * S * D;
  float* voutp = koutp + (size_t)B * S * D;

  // 1) weight fake-quant -> f16
  k_quant_weight<<<N_QKV, 256, 0, stream>>>(w_attn, wq16, D);
  k_quant_weight<<<D,     256, 0, stream>>>(w_proj, wp16, D);
  // 2) activations -> f16
  k_f32_to_f16<<<4096, 256, 0, stream>>>(hidden, hidden16, TOK * D);
  // 3) QKV GEMM (WMMA + async global->LDS weight staging)
  {
    int blocks = (TOK / 128) * (N_QKV / 64);        // 32 * 48 = 1536
    k_gemm_nt_wmma<<<blocks, 256, 0, stream>>>(hidden16, wq16, b_attn, qkv32,
                                               TOK, N_QKV, D);
  }
  // 4) KV fake-quant scales + head split (k,v outputs land in d_out)
  k_kv_amax<<<256, 256, 0, stream>>>(qkv32, kamax, vamax);
  k_build_heads<<<4096, 256, 0, stream>>>(qkv32, kamax, vamax, q16, k16, vT16,
                                          koutp, voutp);
  // 5) causal flash attention (WMMA)
  {
    int waves = B * H * (S / 16);          // 4096
    k_flash_attn<<<waves / 4, 128, 0, stream>>>(q16, k16, vT16, attnout);
  }
  // 6) per-batch activation fake-quant -> f16
  k_act_partial<<<B * 256, 256, 0, stream>>>(attnout, part);
  k_act_reduce<<<B, 256, 0, stream>>>(part, actamax);
  k_act_quant<<<4096, 256, 0, stream>>>(attnout, actamax, attnq16);
  // 7) output projection GEMM (WMMA) -> d_out
  {
    int blocks = (TOK / 128) * (D / 64);            // 32 * 16 = 512
    k_gemm_nt_wmma<<<blocks, 256, 0, stream>>>(attnq16, wp16, b_proj, outp,
                                               TOK, D, D);
  }
}
